// EdgeAwareRefinement_27238682591979
// MI455X (gfx1250) — matrix operand order repaired and checked by reading, not compiled, not run on hardware
//
#include <hip/hip_runtime.h>

typedef __attribute__((ext_vector_type(16))) _Float16 v16h;
typedef __attribute__((ext_vector_type(8)))  _Float16 v8h;
typedef __attribute__((ext_vector_type(8)))  float    v8f;

static constexpr int B  = 8;
static constexpr int H  = 512;
static constexpr int W  = 640;
static constexpr int HW = H * W;

// ---------------------------------------------------------------------------
// conv1: (edges|inputs) 4ch -> 32ch, 3x3, +bias, ReLU.
// Output x1 in NHWC f16 (ci fastest) so each pixel is a contiguous 64B record:
// coalesced vector stores here, contiguous/async-friendly loads in conv2.
// ---------------------------------------------------------------------------
__global__ __launch_bounds__(256)
void conv1_kernel(const float* __restrict__ edges, const float* __restrict__ inputs,
                  const float* __restrict__ w1, const float* __restrict__ b1,
                  _Float16* __restrict__ x1)
{
    __shared__ float sW[32 * 36];   // matches w1 (32,4,3,3) flat: co*36 + ci*9 + tap
    __shared__ float sB[32];
    for (int i = threadIdx.x; i < 32 * 36; i += 256) sW[i] = w1[i];
    if (threadIdx.x < 32) sB[threadIdx.x] = b1[threadIdx.x];
    __syncthreads();

    int gid = blockIdx.x * 256 + threadIdx.x;     // over B*H*W exactly
    int b = gid / HW;
    int p = gid - b * HW;
    int y = p / W;
    int x = p - y * W;

    float in[36];
#pragma unroll
    for (int ci = 0; ci < 4; ++ci) {
        const float* src = (ci == 0) ? (edges + (size_t)b * HW)
                                     : (inputs + ((size_t)b * 3 + (ci - 1)) * HW);
#pragma unroll
        for (int tap = 0; tap < 9; ++tap) {
            int yy = y + tap / 3 - 1;
            int xx = x + tap % 3 - 1;
            in[ci * 9 + tap] = ((unsigned)yy < (unsigned)H && (unsigned)xx < (unsigned)W)
                                   ? src[yy * W + xx] : 0.f;
        }
    }

    v16h o0, o1;
#pragma unroll
    for (int co = 0; co < 32; ++co) {
        float acc = sB[co];
#pragma unroll
        for (int j = 0; j < 36; ++j) acc = fmaf(sW[co * 36 + j], in[j], acc);
        acc = acc > 0.f ? acc : 0.f;
        if (co < 16) o0[co] = (_Float16)acc; else o1[co - 16] = (_Float16)acc;
    }
    _Float16* ob = x1 + ((size_t)b * HW + p) * 32;   // NHWC record, 64B aligned
    *(v16h*)&ob[0]  = o0;
    *(v16h*)&ob[16] = o1;
}

// ---------------------------------------------------------------------------
// Fused conv2 (+ReLU) -> convp (+edges, clip). x2 never touches global memory.
// Block: 256 thr (8 waves); output: ew tile 8y x 32x (8 channels, f32).
//   conv2 computes x2 on a 10y x 34x haloed region in LDS (30 row/tile pairs,
//   each: 9 taps x 2 co-tiles of v_wmma_f32_16x16x32_f16).
//   convp consumes the LDS x2 tile (16 pairs, 9 wmma each).
// x1 tile (12x36 px, NHWC) staged with CDNA5 async copies to LDS (ASYNCcnt).
// ---------------------------------------------------------------------------
__global__ __launch_bounds__(256)
void conv2p_fused_kernel(const _Float16* __restrict__ x1,   // NHWC (B,H,W,32) f16
                         const float* __restrict__ w2, const float* __restrict__ b2,
                         const float* __restrict__ wp, const float* __restrict__ bp,
                         const float* __restrict__ edges, float* __restrict__ ew)
{
    __shared__ __align__(32) _Float16 sX1[12 * 36 * 32]; // [ty][tx][ci]  27.6 KB
    __shared__ __align__(32) _Float16 sX2[10 * 34 * 32]; // [ty][tx][ci]  21.8 KB
    __shared__ __align__(32) _Float16 sW2[9 * 32 * 32];  // [tap][co][ci] 18.4 KB
    __shared__ __align__(32) _Float16 sWP[9 * 16 * 32];  // [tap][co][ci]  9.2 KB

    const int x0 = blockIdx.x * 32;
    const int y0 = blockIdx.y * 8;
    const int bb = blockIdx.z;
    const int t  = threadIdx.x;

    // ---- stage weights (f32 -> f16) ----
    for (int i = t; i < 9 * 32 * 32; i += 256) {
        int tap = i / 1024, rem = i - tap * 1024;
        int co = rem >> 5, ci = rem & 31;
        sW2[i] = (_Float16)w2[(co * 32 + ci) * 9 + tap];
    }
    for (int i = t; i < 9 * 16 * 32; i += 256) {
        int tap = i / 512, rem = i - tap * 512;
        int co = rem >> 5, ci = rem & 31;
        sWP[i] = (co < 8) ? (_Float16)wp[(co * 32 + ci) * 9 + tap] : (_Float16)0.f;
    }

    // ---- async-stage x1 tile: 12x36 px, each px = 64B (4 x 16B chunks) ----
    const _Float16* xb = x1 + (size_t)bb * HW * 32;
    const uint32_t ldsbase = (uint32_t)(uintptr_t)sX1;   // wave-relative LDS offset
    for (int c = t; c < 432 * 4; c += 256) {             // 1728 16B chunks
        int p  = c >> 2, q = c & 3;
        int ty = p / 36, tx = p - ty * 36;
        int gy = y0 - 2 + ty, gx = x0 - 2 + tx;
        if ((unsigned)gy < (unsigned)H && (unsigned)gx < (unsigned)W) {
            uint32_t dst = ldsbase + (uint32_t)(p * 32 + q * 8) * 2;
            uint64_t ga  = (uint64_t)(uintptr_t)(xb + ((size_t)gy * W + gx) * 32 + q * 8);
            asm volatile("global_load_async_to_lds_b128 %0, %1, off"
                         :: "v"(dst), "v"(ga) : "memory");
        } else {
            v8h z = {};
            *(v8h*)&sX1[p * 32 + q * 8] = z;             // zero halo
        }
    }
    asm volatile("s_wait_asynccnt 0x0" ::: "memory");
    __syncthreads();

    const int wave = t >> 5;
    const int lane = t & 31;
    const int n    = lane & 15;      // px col in B/C/D; row m in A
    const int half = lane >> 4;

    // ---- phase 1: conv2 into sX2 (x2 rows 0..9 of 34 px, 32 co, ReLU) ----
    for (int it = 0; it < 4; ++it) {
        int pi = wave + 8 * it;                  // wave-uniform; EXEC stays full
        if (pi < 30) {
            int r  = pi / 3;                     // x2 row 0..9
            int j  = pi - r * 3;                 // px tile 0..2
            int tx0 = (j == 2) ? 18 : j * 16;    // tiles [0,16), [16,32), [18,34)
            v8f c0 = {}; v8f c1 = {};
#pragma unroll
            for (int tap = 0; tap < 9; ++tap) {
                const int ky = tap / 3, kx = tap % 3;
                const v16h bf = *(const v16h*)&sX1[((r + ky) * 36 + tx0 + n + kx) * 32 + half * 16];
                const int ab0 = (tap * 32 + n) * 32 + half * 16;
                v8h a0lo = *(const v8h*)&sW2[ab0];
                v8h a0hi = *(const v8h*)&sW2[ab0 + 8];
                v16h a0 = __builtin_shufflevector(a0lo, a0hi, 0,1,2,3,4,5,6,7,8,9,10,11,12,13,14,15);
                const int ab1 = (tap * 32 + 16 + n) * 32 + half * 16;
                v8h a1lo = *(const v8h*)&sW2[ab1];
                v8h a1hi = *(const v8h*)&sW2[ab1 + 8];
                v16h a1 = __builtin_shufflevector(a1lo, a1hi, 0,1,2,3,4,5,6,7,8,9,10,11,12,13,14,15);
                c0 = __builtin_amdgcn_wmma_f32_16x16x32_f16(false, a0, false, bf, (short)0, c0, false, false);
                c1 = __builtin_amdgcn_wmma_f32_16x16x32_f16(false, a1, false, bf, (short)0, c1, false, false);
            }
            v8h o0, o1;
#pragma unroll
            for (int rr = 0; rr < 8; ++rr) {
                float v0 = c0[rr] + b2[rr + half * 8];
                float v1 = c1[rr] + b2[rr + half * 8 + 16];
                o0[rr] = (_Float16)(v0 > 0.f ? v0 : 0.f);
                o1[rr] = (_Float16)(v1 > 0.f ? v1 : 0.f);
            }
            const int xo = (r * 34 + tx0 + n) * 32;
            *(v8h*)&sX2[xo + half * 8]      = o0;        // co 0..15 half
            *(v8h*)&sX2[xo + 16 + half * 8] = o1;        // co 16..31 half
        }
    }
    __syncthreads();

    // ---- phase 2: convp + edges + clip -> ew (B,8,H,W) f32 ----
    for (int it = 0; it < 2; ++it) {
        int pi = wave + 8 * it;                  // 0..15
        int r  = pi >> 1;                        // ew row 0..7
        int j  = pi & 1;                         // px tile 0..1
        int tx0 = j * 16;
        v8f c = {};
#pragma unroll
        for (int tap = 0; tap < 9; ++tap) {
            const int ky = tap / 3, kx = tap % 3;
            const v16h bf = *(const v16h*)&sX2[((r + ky) * 34 + tx0 + n + kx) * 32 + half * 16];
            const int ab = (tap * 16 + n) * 32 + half * 16;
            v8h alo = *(const v8h*)&sWP[ab];
            v8h ahi = *(const v8h*)&sWP[ab + 8];
            v16h a = __builtin_shufflevector(alo, ahi, 0,1,2,3,4,5,6,7,8,9,10,11,12,13,14,15);
            c = __builtin_amdgcn_wmma_f32_16x16x32_f16(false, a, false, bf, (short)0, c, false, false);
        }
        if (half == 0) {                         // only co rows 0..7 are valid
            const int gy = y0 + r, gx = x0 + tx0 + n;
            const float e = edges[(size_t)bb * HW + gy * W + gx];
#pragma unroll
            for (int rr = 0; rr < 8; ++rr) {
                float v = c[rr] + bp[rr] + e;
                v = fminf(fmaxf(v, 0.f), 1.f);
                ew[((size_t)bb * 8 + rr) * HW + gy * W + gx] = v;
            }
        }
    }
}

// ---------------------------------------------------------------------------
// All 4 propagate passes fused in one kernel via LDS ping-pong with a
// 1-px-per-direction halo. Tile: 16y x 64x output; d region 18x66.
// ---------------------------------------------------------------------------
__global__ __launch_bounds__(256)
void prop_fused_kernel(const float* __restrict__ depth, const float* __restrict__ ew,
                       float* __restrict__ out)
{
    __shared__ float bufA[18 * 66];
    __shared__ float bufB[18 * 66];
    const int x0 = blockIdx.x * 64;
    const int y0 = blockIdx.y * 16;
    const int bb = blockIdx.z;
    const int t  = threadIdx.x;
    const float* db  = depth + (size_t)bb * HW;
    const float* ewb = ew + (size_t)bb * 8 * HW;

    // depth region rows y0-1..y0+16, cols x0-1..x0+64 (zero outside image)
    for (int i = t; i < 18 * 66; i += 256) {
        int r = i / 66, c = i - r * 66;
        int gy = y0 - 1 + r, gx = x0 - 1 + c;
        bufA[i] = ((unsigned)gy < (unsigned)H && (unsigned)gx < (unsigned)W) ? db[gy * W + gx] : 0.f;
    }
    __syncthreads();

    // pass 1: left->right (s = x-1), ch 0*4 ; rows 0..17, cols 1..65 -> bufB
    for (int i = t; i < 18 * 65; i += 256) {
        int r = i / 65, c = 1 + (i - r * 65);
        int gy = y0 - 1 + r, gx = x0 - 1 + c;
        float v = 0.f;
        if ((unsigned)gy < (unsigned)H && (unsigned)gx < (unsigned)W) {
            float w = ewb[0 * HW + gy * W + gx] * ewb[4 * HW + gy * W + gx];
            v = bufA[r * 66 + c - 1] * w + bufA[r * 66 + c] * (1.f - w);
        }
        bufB[r * 66 + c] = v;
    }
    __syncthreads();

    // pass 2: right->left (s = x+1), ch 1*5 ; rows 0..17, cols 1..64 -> bufA
    for (int i = t; i < 18 * 64; i += 256) {
        int r = i / 64, c = 1 + (i - r * 64);
        int gy = y0 - 1 + r, gx = x0 - 1 + c;
        float v = 0.f;
        if ((unsigned)gy < (unsigned)H && (unsigned)gx < (unsigned)W) {
            float w = ewb[1 * HW + gy * W + gx] * ewb[5 * HW + gy * W + gx];
            v = bufB[r * 66 + c + 1] * w + bufB[r * 66 + c] * (1.f - w);
        }
        bufA[r * 66 + c] = v;
    }
    __syncthreads();

    // pass 3: up->down (s = y-1), ch 2*6 ; rows 1..17, cols 1..64 -> bufB
    for (int i = t; i < 17 * 64; i += 256) {
        int r = 1 + i / 64, c = 1 + (i & 63);
        int gy = y0 - 1 + r, gx = x0 - 1 + c;
        float v = 0.f;
        if ((unsigned)gy < (unsigned)H && (unsigned)gx < (unsigned)W) {
            float w = ewb[2 * HW + gy * W + gx] * ewb[6 * HW + gy * W + gx];
            v = bufA[(r - 1) * 66 + c] * w + bufA[r * 66 + c] * (1.f - w);
        }
        bufB[r * 66 + c] = v;
    }
    __syncthreads();

    // pass 4: down->up (s = y+1), ch 3*7 ; rows 1..16, cols 1..64 -> out
    for (int i = t; i < 16 * 64; i += 256) {
        int r = 1 + i / 64, c = 1 + (i & 63);
        int gy = y0 - 1 + r, gx = x0 - 1 + c;          // always in-image
        float w = ewb[3 * HW + gy * W + gx] * ewb[7 * HW + gy * W + gx];
        out[(size_t)bb * HW + gy * W + gx] =
            bufB[(r + 1) * 66 + c] * w + bufB[r * 66 + c] * (1.f - w);
    }
}

// ---------------------------------------------------------------------------
extern "C" void kernel_launch(void* const* d_in, const int* in_sizes, int n_in,
                              void* d_out, int out_size, void* d_ws, size_t ws_size,
                              hipStream_t stream)
{
    (void)in_sizes; (void)n_in; (void)out_size; (void)ws_size;
    const float* inputs = (const float*)d_in[0];   // (8,3,512,640)
    const float* edges  = (const float*)d_in[1];   // (8,1,512,640)
    const float* depth  = (const float*)d_in[2];   // (8,1,512,640)
    const float* w1     = (const float*)d_in[3];   // (32,4,3,3)
    const float* b1     = (const float*)d_in[4];
    const float* w2     = (const float*)d_in[5];   // (32,32,3,3)
    const float* b2     = (const float*)d_in[6];
    const float* wp     = (const float*)d_in[7];   // (8,32,3,3)
    const float* bp     = (const float*)d_in[8];

    char* ws = (char*)d_ws;
    _Float16* x1 = (_Float16*)ws;                                  // NHWC f16, 167.8 MB
    float*    ew = (float*)(ws + (size_t)B * HW * 32 * sizeof(_Float16)); // 83.9 MB

    conv1_kernel<<<(B * HW) / 256, 256, 0, stream>>>(edges, inputs, w1, b1, x1);
    conv2p_fused_kernel<<<dim3(W / 32, H / 8, B), 256, 0, stream>>>(x1, w2, b2, wp, bp, edges, ew);
    prop_fused_kernel<<<dim3(W / 64, H / 16, B), 256, 0, stream>>>(depth, ew, (float*)d_out);
}
